// IDANSE_19052474925189
// MI455X (gfx1250) — matrix operand order repaired and checked by reading, not compiled, not run on hardware
//
#include <hip/hip_runtime.h>
#include <hip/hip_bf16.h>
#include <stddef.h>

// ---------------------------------------------------------------------------
// IDANSE likelihood on gfx1250 (CDNA5, wave32, WMMA 16x16x32 f16)
//   N=256 T=1000 NO=10 NS=10 HID=64 D=20 3H=192
// All WMMA operands are kept in LDS in fragment-native packed layout so each
// fragment load is two ds_load_b128 (32 contiguous bytes per lane).
// ---------------------------------------------------------------------------

#define NB    256
#define TT    1000
#define NOB   10
#define NSD   10
#define HID   64
#define DIN   20
#define H3    192
#define NT12  12      // 192/16 column tiles

typedef __attribute__((ext_vector_type(16))) _Float16 v16h;
typedef __attribute__((ext_vector_type(8)))  float    v8f;

#define WMMA_F16F32(a, b, c) \
  __builtin_amdgcn_wmma_f32_16x16x32_f16(false, (a), false, (b), (short)0, (c), false, false)

// A fragment element mapping (16-bit A, 16x32): lane l, elem e ->
//   M = l&15,  K = 8*(l>>4) + (e&7) + (e>=8 ? 16 : 0)
// Inverse (given k in [0,32)): half = (k>>3)&1, e = (k&7) + ((k&16) ? 8 : 0)
// B fragment element mapping (16-bit B, 32x16): lane l, elem e ->
//   N = l&15,  K = 8*(l>>4) + (e&7) + (e>=8 ? 16 : 0)

// ---------------------------------------------------------------------------
// Kernel 1: persistent GRU over time. 16 blocks x 384 threads (12 waves).
// ---------------------------------------------------------------------------
__global__ __launch_bounds__(384) void idanse_gru_fwd(
    const float* __restrict__ Yi, const float* __restrict__ Xh,
    const float* __restrict__ Wi, const float* __restrict__ Wh,
    const float* __restrict__ bi, const float* __restrict__ bh,
    const float* __restrict__ Wmu, const float* __restrict__ bmu,
    const float* __restrict__ Wvar, const float* __restrict__ bvar,
    float* __restrict__ mu_ws, float* __restrict__ var_ws) {
  // B fragments, packed: [tile][lane][16]  (Wi: K padded 20->32, one chunk)
  __shared__ _Float16 sWiB[NT12 * 32 * 16];          // 12 KB
  // [tile][chunk(2)][lane][16]
  __shared__ _Float16 sWhB[NT12 * 2 * 32 * 16];      // 24 KB
  __shared__ float    sbi[H3], sbh[H3];
  __shared__ float    sWmu[HID * NSD], sWvar[HID * NSD];
  __shared__ float    sbmu[NSD], sbvar[NSD];
  __shared__ float    h32[16 * HID];                 // f32 hidden state
  __shared__ _Float16 h16A[2 * 32 * 16];             // h in A-fragment layout (2 K-chunks)
  __shared__ _Float16 xA[32 * 16];                   // x tile in A-fragment layout
  __shared__ float    xp[16 * H3];                   // x@Wi+bi
  __shared__ float    hp[16 * H3];                   // h@Wh+bh

  const int tid  = threadIdx.x;
  const int lane = tid & 31;
  const int wave = tid >> 5;                         // 0..11
  const int n0   = blockIdx.x * 16;

  // ---- pack Wi into B-fragment layout (zero-padded K 20..31) ----
  for (int i = tid; i < NT12 * 32 * 16; i += 384) {
    const int ct = i >> 9, l = (i >> 4) & 31, e = i & 15;
    const int n = l & 15, half = l >> 4;
    const int k = (half << 3) + (e & 7) + ((e >= 8) ? 16 : 0);   // 0..31
    sWiB[i] = (k < DIN) ? (_Float16)Wi[k * H3 + ct * 16 + n] : (_Float16)0.0f;
  }
  // ---- pack Wh into B-fragment layout (two K-chunks) ----
  for (int i = tid; i < NT12 * 2 * 32 * 16; i += 384) {
    const int ct = i >> 10, cb = (i >> 9) & 1, l = (i >> 4) & 31, e = i & 15;
    const int n = l & 15, half = l >> 4;
    const int k = (cb << 5) + (half << 3) + (e & 7) + ((e >= 8) ? 16 : 0);   // 0..63
    sWhB[i] = (_Float16)Wh[k * H3 + ct * 16 + n];
  }
  for (int i = tid; i < H3; i += 384) { sbi[i] = bi[i]; sbh[i] = bh[i]; }
  for (int i = tid; i < HID * NSD; i += 384) { sWmu[i] = Wmu[i]; sWvar[i] = Wvar[i]; }
  if (tid < NSD) { sbmu[tid] = bmu[tid]; sbvar[tid] = bvar[tid]; }
  for (int i = tid; i < 16 * HID; i += 384) h32[i] = 0.0f;
  for (int i = tid; i < 2 * 32 * 16; i += 384) h16A[i] = (_Float16)0.0f;
  __syncthreads();

  for (int t = 0; t < TT; ++t) {
    // ---- stage x_t directly in A-fragment layout (K: 0..9=Yi, 10..19=Xh, 20..31=0)
    for (int i = tid; i < 32 * 16; i += 384) {
      const int l = i >> 4, e = i & 15;
      const int m = l & 15, half = l >> 4;
      const int k = (half << 3) + (e & 7) + ((e >= 8) ? 16 : 0);
      float v = 0.0f;
      if (k < NOB)      v = Yi[((size_t)(n0 + m) * TT + t) * NOB + k];
      else if (k < DIN) v = Xh[((size_t)(n0 + m) * TT + t) * NSD + (k - NOB)];
      xA[i] = (_Float16)v;
    }
    if (t + 1 < TT && tid < 16) {   // global_prefetch_b8 for next step
      __builtin_prefetch(&Yi[((size_t)(n0 + tid) * TT + (t + 1)) * NOB], 0, 1);
      __builtin_prefetch(&Xh[((size_t)(n0 + tid) * TT + (t + 1)) * NSD], 0, 1);
    }
    __syncthreads();

    // ---- WMMA: each wave owns 16 output columns; all operands are packed ----
    {
      const int c0   = wave * 16;
      const int ncol = lane & 15, half = lane >> 4;
      const float bx = sbi[c0 + ncol];
      const float bz = sbh[c0 + ncol];
      v8f cx = {bx, bx, bx, bx, bx, bx, bx, bx};     // bias is per-column: exact splat
      v8f ch = {bz, bz, bz, bz, bz, bz, bz, bz};

      const v16h ax  = *(const v16h*)(xA + lane * 16);
      const v16h bwi = *(const v16h*)(sWiB + (wave << 9) + lane * 16);
      cx = WMMA_F16F32(ax, bwi, cx);

      const v16h ah0 = *(const v16h*)(h16A + lane * 16);
      const v16h ah1 = *(const v16h*)(h16A + 512 + lane * 16);
      const v16h bw0 = *(const v16h*)(sWhB + (wave << 10) + lane * 16);
      const v16h bw1 = *(const v16h*)(sWhB + (wave << 10) + 512 + lane * 16);
      ch = WMMA_F16F32(ah0, bw0, ch);
      ch = WMMA_F16F32(ah1, bw1, ch);

#pragma unroll
      for (int v = 0; v < 8; ++v) {
        const int row = v + 8 * half;   // C layout: M = v + 8*(lane>>4), N = lane&15
        xp[row * H3 + c0 + ncol] = cx[v];
        hp[row * H3 + c0 + ncol] = ch[v];
      }
    }
    __syncthreads();

    // ---- gates: r=sig(xr+hr), z=sig(xz+hz), n=tanh(xn+r*hn) ----
    for (int i = tid; i < 16 * HID; i += 384) {
      const int r = i >> 6, j = i & 63;
      const float xr = xp[r * H3 + j],       hr = hp[r * H3 + j];
      const float xz = xp[r * H3 + 64 + j],  hz = hp[r * H3 + 64 + j];
      const float xn = xp[r * H3 + 128 + j], hn = hp[r * H3 + 128 + j];
      const float rg = 1.0f / (1.0f + expf(-(xr + hr)));
      const float zg = 1.0f / (1.0f + expf(-(xz + hz)));
      const float ng = tanhf(xn + rg * hn);
      const float hnew = (1.0f - zg) * ng + zg * h32[r * HID + j];
      h32[r * HID + j] = hnew;
      // write h16 directly in A-fragment layout for next step's WMMA
      const int chunk = j >> 5, kp = j & 31;
      const int ah = (kp >> 3) & 1;
      const int ae = (kp & 7) + ((kp & 16) ? 8 : 0);
      h16A[(chunk << 9) + ((r + (ah << 4)) << 4) + ae] = (_Float16)hnew;
    }
    __syncthreads();

    // ---- heads: mu = h@Wmu+bmu, var = softplus(h@Wvar+bvar) ----
    if (tid < 16 * 2 * NSD) {
      const int r = tid / (2 * NSD);
      const int q = tid % (2 * NSD);
      const int c = q % NSD;
      const bool isvar = (q >= NSD);
      const float* W = isvar ? sWvar : sWmu;
      float acc = isvar ? sbvar[c] : sbmu[c];
#pragma unroll
      for (int j = 0; j < HID; ++j) acc += h32[r * HID + j] * W[j * NSD + c];
      const size_t o = ((size_t)(n0 + r) * TT + t) * NSD + c;
      if (isvar) var_ws[o] = (acc > 20.0f) ? acc : log1pf(expf(acc));
      else       mu_ws[o] = acc;
    }
    __syncthreads();
  }
}

// ---------------------------------------------------------------------------
// Kernel 2: per-(n,t) Gaussian log-likelihood term via 10x10 Cholesky.
// ---------------------------------------------------------------------------
__global__ __launch_bounds__(256) void idanse_loglik(
    const float* __restrict__ Yi, const float* __restrict__ Cw,
    const float* __restrict__ Hm, const float* __restrict__ mu_w,
    const float* __restrict__ mu_ws, const float* __restrict__ var_ws,
    float* __restrict__ contrib) {
  __shared__ float sH[NOB * NSD];
  __shared__ float smw[NOB];
  const int tid = threadIdx.x;
  if (tid < NOB * NSD) sH[tid] = Hm[tid];
  if (tid < NOB) smw[tid] = mu_w[tid];
  __syncthreads();

  const int gid = blockIdx.x * 256 + tid;         // 0 .. N*T-1
  const int n = gid / TT;
  const size_t base = (size_t)gid * NSD;

  float mu[NSD], var[NSD], e[NOB];
#pragma unroll
  for (int i = 0; i < NSD; ++i) { mu[i] = mu_ws[base + i]; var[i] = var_ws[base + i]; }
#pragma unroll
  for (int i = 0; i < NOB; ++i) {
    float m = smw[i];
#pragma unroll
    for (int j = 0; j < NSD; ++j) m += sH[i * NSD + j] * mu[j];
    e[i] = Yi[base + i] - m;
  }

  // A = H diag(var) H^T + Cw[n]  (lower triangle only)
  float A[NOB][NOB];
  const float* Cwn = Cw + (size_t)n * NOB * NOB;
#pragma unroll
  for (int i = 0; i < NOB; ++i)
#pragma unroll
    for (int k = 0; k <= i; ++k) {
      float s = 0.0f;
#pragma unroll
      for (int j = 0; j < NSD; ++j) s += sH[i * NSD + j] * var[j] * sH[k * NSD + j];
      A[i][k] = s + Cwn[i * NOB + k];
    }

  // in-place lower Cholesky
  float logdet = 0.0f;
#pragma unroll
  for (int i = 0; i < NOB; ++i) {
#pragma unroll
    for (int j = 0; j <= i; ++j) {
      float s = A[i][j];
#pragma unroll
      for (int q = 0; q < j; ++q) s -= A[i][q] * A[j][q];
      if (j == i) { const float d = sqrtf(s); A[i][i] = d; logdet += logf(d); }
      else         A[i][j] = s / A[j][j];
    }
  }
  logdet *= 2.0f;

  // solve L y = e ; L^T x = y ; quad = e . x
  float y[NOB];
#pragma unroll
  for (int i = 0; i < NOB; ++i) {
    float s = e[i];
#pragma unroll
    for (int j = 0; j < i; ++j) s -= A[i][j] * y[j];
    y[i] = s / A[i][i];
  }
  float x[NOB];
#pragma unroll
  for (int i = NOB - 1; i >= 0; --i) {
    float s = y[i];
#pragma unroll
    for (int j = i + 1; j < NOB; ++j) s -= A[j][i] * x[j];
    x[i] = s / A[i][i];
  }
  float quad = 0.0f;
#pragma unroll
  for (int i = 0; i < NOB; ++i) quad += e[i] * x[i];

  contrib[gid] = logdet + quad;
}

// ---------------------------------------------------------------------------
// Kernel 3: deterministic reduction -> scalar.
//   out = -0.5*log(2*pi) - 0.5 * sum(contrib) / (N*T*NO)
// ---------------------------------------------------------------------------
__global__ __launch_bounds__(256) void idanse_reduce(
    const float* __restrict__ contrib, float* __restrict__ out) {
  __shared__ float s[256];
  const int tid = threadIdx.x;
  float acc = 0.0f;
  for (int i = tid; i < NB * TT; i += 256) acc += contrib[i];
  s[tid] = acc;
  __syncthreads();
  for (int w = 128; w > 0; w >>= 1) {
    if (tid < w) s[tid] += s[tid + w];
    __syncthreads();
  }
  if (tid == 0) {
    const float log2pi = 1.8378770664093453f;
    out[0] = -0.5f * log2pi - 0.5f * s[0] / ((float)NB * (float)TT * (float)NOB);
  }
}

extern "C" void kernel_launch(void* const* d_in, const int* in_sizes, int n_in,
                              void* d_out, int out_size, void* d_ws, size_t ws_size,
                              hipStream_t stream) {
  (void)in_sizes; (void)n_in; (void)out_size; (void)ws_size;
  const float* Yi   = (const float*)d_in[0];
  const float* Xh   = (const float*)d_in[1];
  const float* Cw   = (const float*)d_in[2];
  const float* Hm   = (const float*)d_in[3];
  const float* mu_w = (const float*)d_in[4];
  const float* Wi   = (const float*)d_in[5];
  const float* Wh   = (const float*)d_in[6];
  const float* bi   = (const float*)d_in[7];
  const float* bh   = (const float*)d_in[8];
  const float* Wmu  = (const float*)d_in[9];
  const float* bmu  = (const float*)d_in[10];
  const float* Wvar = (const float*)d_in[11];
  const float* bvar = (const float*)d_in[12];

  float* ws      = (float*)d_ws;
  float* mu_ws   = ws;                               // N*T*NS floats
  float* var_ws  = ws + (size_t)NB * TT * NSD;       // N*T*NS floats
  float* contrib = ws + (size_t)2 * NB * TT * NSD;   // N*T floats

  idanse_gru_fwd<<<NB / 16, 384, 0, stream>>>(Yi, Xh, Wi, Wh, bi, bh,
                                              Wmu, bmu, Wvar, bvar, mu_ws, var_ws);
  idanse_loglik<<<(NB * TT) / 256, 256, 0, stream>>>(Yi, Cw, Hm, mu_w,
                                                     mu_ws, var_ws, contrib);
  idanse_reduce<<<1, 256, 0, stream>>>(contrib, (float*)d_out);
}
